// Enc_block_33182917329086
// MI455X (gfx1250) — matrix operand order repaired and checked by reading, not compiled, not run on hardware
//
#include <hip/hip_runtime.h>
#include <algorithm>

typedef _Float16 f16;
typedef __attribute__((ext_vector_type(16))) _Float16 v16h;
typedef __attribute__((ext_vector_type(8)))  float    v8f;

#define INF_F 3.402823466e+38f

// ---------------------------------------------------------------------------
// Pre-swizzle B[K x Nact] (row-major fp32) into WMMA B-fragment order (f16),
// zero-padded to Npad columns. Fragment layout (per 16x16 n-tile, per 32-K
// chunk): element (nt, kc, lane, e) holds B[kc*32 + (lane>>4)*16 + e,
// nt*16 + (lane&15)] so each lane's v16h is one contiguous 32-byte chunk.
// Also pads the bias vector (zeros where col >= Nact or bias == null).
// ---------------------------------------------------------------------------
__global__ void eb_swizzleB(const float* __restrict__ B, const float* __restrict__ bias,
                            int K, int Nact, int Npad, int ldb,
                            f16* __restrict__ Bsw, float* __restrict__ biasPad)
{
    const int kChunks = K >> 5;
    const int total = (Npad >> 4) * kChunks * 32 * 16;
    const int stride = gridDim.x * blockDim.x;
    for (int t = blockIdx.x * blockDim.x + threadIdx.x; t < total; t += stride) {
        const int e    = t & 15;
        const int lane = (t >> 4) & 31;
        const int rest = t >> 9;
        const int kc   = rest % kChunks;
        const int nt   = rest / kChunks;
        const int k = kc * 32 + (lane >> 4) * 16 + e;
        const int n = nt * 16 + (lane & 15);
        Bsw[t] = (n < Nact) ? (f16)B[(long)k * ldb + n] : (f16)0.0f;
    }
    for (int n = blockIdx.x * blockDim.x + threadIdx.x; n < Npad; n += stride)
        biasPad[n] = (bias && n < Nact) ? bias[n] : 0.0f;
}

// ---------------------------------------------------------------------------
// WMMA GEMM:  D[M x Npad] = A[M x K] @ Bsw (+biasPad)(+residual)
// One 16x64 output strip per wave: A fragment loaded once, 4 WMMAs.
// M mult of 16, K mult of 32, Npad mult of 64. No guards -> EXEC stays all-1s.
// ---------------------------------------------------------------------------
__global__ void eb_gemm_wmma(const float* __restrict__ A, const f16* __restrict__ Bsw,
                             const float* __restrict__ biasPad, const float* __restrict__ residual,
                             float* __restrict__ D, int M, int Npad, int K, int lda, int ldd)
{
    const int wavesPerBlock = blockDim.x >> 5;
    const int wave = blockIdx.x * wavesPerBlock + (threadIdx.x >> 5);
    const int lane = threadIdx.x & 31;
    const int lhalf = lane >> 4;
    const int l15   = lane & 15;
    const int groupsN = Npad >> 6;
    const int nTasks = (M >> 4) * groupsN;
    const int kChunks = K >> 5;
    const int waveStride = gridDim.x * wavesPerBlock;
    const v16h* __restrict__ Bv = (const v16h*)Bsw;

    for (int t = wave; t < nTasks; t += waveStride) {
        const int mt = t / groupsN;
        const int ng = t - mt * groupsN;
        const int m0 = mt << 4;
        const int n0 = ng << 6;

        v8f acc0 = {}, acc1 = {}, acc2 = {}, acc3 = {};
        const float* __restrict__ Arow = A + (long)(m0 + l15) * lda;

        for (int kc = 0; kc < kChunks; ++kc) {
            const int k0 = kc << 5;
            // A fragment (16x32): lane-half picks contiguous K range {lh*16..lh*16+15}
            v16h a;
            const float* __restrict__ ap0 = Arow + k0 + lhalf * 16;
            const float* __restrict__ ap1 = Arow + k0 + lhalf * 16 + 8;
            #pragma unroll
            for (int e = 0; e < 8; ++e) a[e] = (f16)ap0[e];
            #pragma unroll
            for (int e = 0; e < 8; ++e) a[8 + e] = (f16)ap1[e];

            // B fragments: contiguous 32B per lane, pre-swizzled
            const long bbase = ((long)(ng * 4) * kChunks + kc) * 32 + lane;
            const v16h b0 = Bv[bbase];
            const v16h b1 = Bv[bbase + (long)kChunks * 32];
            const v16h b2 = Bv[bbase + (long)kChunks * 64];
            const v16h b3 = Bv[bbase + (long)kChunks * 96];

            acc0 = __builtin_amdgcn_wmma_f32_16x16x32_f16(false, a, false, b0, (short)0, acc0, false, false);
            acc1 = __builtin_amdgcn_wmma_f32_16x16x32_f16(false, a, false, b1, (short)0, acc1, false, false);
            acc2 = __builtin_amdgcn_wmma_f32_16x16x32_f16(false, a, false, b2, (short)0, acc2, false, false);
            acc3 = __builtin_amdgcn_wmma_f32_16x16x32_f16(false, a, false, b3, (short)0, acc3, false, false);
        }

        // C/D layout: element r -> row m0 + lhalf*8 + r, col = tile*16 + l15
        const int colBase = n0 + l15;
        const float bb0 = biasPad[colBase];
        const float bb1 = biasPad[colBase + 16];
        const float bb2 = biasPad[colBase + 32];
        const float bb3 = biasPad[colBase + 48];
        #pragma unroll
        for (int r = 0; r < 8; ++r) {
            const int mrow = m0 + lhalf * 8 + r;
            float* __restrict__ Dr = D + (long)mrow * ldd + colBase;
            float v0 = acc0[r] + bb0;
            float v1 = acc1[r] + bb1;
            float v2 = acc2[r] + bb2;
            float v3 = acc3[r] + bb3;
            if (residual) {
                const float* __restrict__ Rr = residual + (long)mrow * ldd + colBase;
                v0 += Rr[0]; v1 += Rr[16]; v2 += Rr[32]; v3 += Rr[48];
            }
            Dr[0] = v0; Dr[16] = v1; Dr[32] = v2; Dr[48] = v3;
        }
    }
}

// ---------------------------------------------------------------------------
// BatchNorm (training mode, biased var): per-column mean & inverse std.
// ---------------------------------------------------------------------------
__global__ void eb_bn_stats(const float* __restrict__ X, int R, int C,
                            float* __restrict__ mean, float* __restrict__ istd)
{
    const int c = blockIdx.x;
    float s = 0.f, s2 = 0.f;
    for (int r = threadIdx.x; r < R; r += blockDim.x) {
        const float v = X[(long)r * C + c];
        s += v; s2 += v * v;
    }
    __shared__ float sh[256], sh2[256];
    sh[threadIdx.x] = s; sh2[threadIdx.x] = s2;
    __syncthreads();
    for (int st = 128; st > 0; st >>= 1) {
        if (threadIdx.x < st) { sh[threadIdx.x] += sh[threadIdx.x + st]; sh2[threadIdx.x] += sh2[threadIdx.x + st]; }
        __syncthreads();
    }
    if (threadIdx.x == 0) {
        const float m = sh[0] / (float)R;
        const float var = sh2[0] / (float)R - m * m;
        mean[c] = m;
        istd[c] = rsqrtf(var + 1e-5f);
    }
}

__global__ void eb_bn_apply_relu(float* __restrict__ X, int total, int C,
                                 const float* __restrict__ mean, const float* __restrict__ istd,
                                 const float* __restrict__ gamma, const float* __restrict__ beta)
{
    for (int t = blockIdx.x * blockDim.x + threadIdx.x; t < total; t += gridDim.x * blockDim.x) {
        const int c = t % C;
        const float v = (X[t] - mean[c]) * istd[c] * gamma[c] + beta[c];
        X[t] = fmaxf(v, 0.0f);
    }
}

// ---------------------------------------------------------------------------
// Neighbor max-pool: pooled pre-init to h (self loop); post-ReLU values >= 0
// so uint-bit atomicMax is order preserving -> native GLOBAL_ATOMIC_MAX_U32.
// ---------------------------------------------------------------------------
__global__ void eb_edge_max(const float* __restrict__ h, const int* __restrict__ ei,
                            int NE, int C, float* __restrict__ pooled)
{
    const int tot = NE * C;
    for (int t = blockIdx.x * blockDim.x + threadIdx.x; t < tot; t += gridDim.x * blockDim.x) {
        const int e = t / C, c = t - e * C;
        const int src = ei[e];
        const int dst = ei[NE + e];
        const float val = h[(long)src * C + c];
        atomicMax((unsigned int*)(pooled + (long)dst * C + c), __float_as_uint(val));
    }
}

// ---------------------------------------------------------------------------
// Farthest point sampling (serial argmax chain) + bitonic sort of indices.
// ---------------------------------------------------------------------------
__global__ void eb_fps(const float* __restrict__ pos, int n, int m, int* __restrict__ idx_out)
{
    __shared__ float sdist[8192];
    __shared__ float rmax[1024];
    __shared__ int   rarg[1024];
    __shared__ int   lastSel;
    const int tid = threadIdx.x;
    for (int i = tid; i < n; i += 1024) sdist[i] = INF_F;
    if (tid == 0) { lastSel = 0; idx_out[0] = 0; }
    __syncthreads();

    for (int it = 1; it < m; ++it) {
        const int last = lastSel;
        const float lx = pos[last * 3 + 0], ly = pos[last * 3 + 1], lz = pos[last * 3 + 2];
        float bm = -1.0f; int ba = tid;
        for (int i = tid; i < n; i += 1024) {
            const float dx = pos[i * 3 + 0] - lx;
            const float dy = pos[i * 3 + 1] - ly;
            const float dz = pos[i * 3 + 2] - lz;
            const float nd = fminf(sdist[i], dx * dx + dy * dy + dz * dz);
            sdist[i] = nd;
            if (nd > bm) { bm = nd; ba = i; }
        }
        rmax[tid] = bm; rarg[tid] = ba;
        __syncthreads();
        for (int s = 512; s > 0; s >>= 1) {
            if (tid < s) {
                const float ov = rmax[tid + s]; const int oi = rarg[tid + s];
                if (ov > rmax[tid] || (ov == rmax[tid] && oi < rarg[tid])) { rmax[tid] = ov; rarg[tid] = oi; }
            }
            __syncthreads();
        }
        if (tid == 0) { lastSel = rarg[0]; idx_out[it] = rarg[0]; }
        __syncthreads();
    }

    __shared__ int skey[4096];
    for (int i = tid; i < m; i += 1024) skey[i] = idx_out[i];
    __syncthreads();
    for (int ksz = 2; ksz <= m; ksz <<= 1) {
        for (int j = ksz >> 1; j > 0; j >>= 1) {
            for (int i = tid; i < m; i += 1024) {
                const int ixj = i ^ j;
                if (ixj > i) {
                    const int a = skey[i], b = skey[ixj];
                    const bool up = ((i & ksz) == 0);
                    if ((up && a > b) || (!up && a < b)) { skey[i] = b; skey[ixj] = a; }
                }
            }
            __syncthreads();
        }
    }
    for (int i = tid; i < m; i += 1024) idx_out[i] = skey[i];
}

__global__ void eb_gather(const float* __restrict__ pooled, const float* __restrict__ pos,
                          const int* __restrict__ idx, int M, int C,
                          float* __restrict__ x1, float* __restrict__ pos1)
{
    const int stride = gridDim.x * blockDim.x;
    for (int t = blockIdx.x * blockDim.x + threadIdx.x; t < M * C; t += stride) {
        const int i = t / C, c = t - i * C;
        x1[t] = pooled[(long)idx[i] * C + c];
    }
    for (int t = blockIdx.x * blockDim.x + threadIdx.x; t < M * 3; t += stride) {
        const int i = t / 3, j = t - i * 3;
        pos1[t] = pos[(long)idx[i] * 3 + j];
    }
}

__global__ void eb_rowsq(const float* __restrict__ x1, int M, int C, float* __restrict__ sq)
{
    for (int i = blockIdx.x * blockDim.x + threadIdx.x; i < M; i += gridDim.x * blockDim.x) {
        float s = 0.f;
        const float* __restrict__ row = x1 + (long)i * C;
        for (int c = 0; c < C; ++c) s += row[c] * row[c];
        sq[i] = s;
    }
}

// ---------------------------------------------------------------------------
// Positional kNN (k=16): LDS-resident distances, 16 argmin passes
// (tie -> lower index, matches lax.top_k). Writes positional edges directly.
// ---------------------------------------------------------------------------
__global__ void eb_knn_pos(const float* __restrict__ pos1, int M,
                           int* __restrict__ s_edges, int* __restrict__ d_edges, int edgeOffset)
{
    const int i = blockIdx.x;
    const int tid = threadIdx.x;
    __shared__ float sd[4096];
    __shared__ float rv[256];
    __shared__ int   ri[256];
    const float px = pos1[i * 3 + 0], py = pos1[i * 3 + 1], pz = pos1[i * 3 + 2];
    for (int j = tid; j < M; j += 256) {
        const float dx = pos1[j * 3 + 0] - px;
        const float dy = pos1[j * 3 + 1] - py;
        const float dz = pos1[j * 3 + 2] - pz;
        sd[j] = (j == i) ? INF_F : (dx * dx + dy * dy + dz * dz);
    }
    __syncthreads();
    for (int k = 0; k < 16; ++k) {
        float bv = INF_F; int bi = M;
        for (int j = tid; j < M; j += 256)
            if (sd[j] < bv) { bv = sd[j]; bi = j; }
        rv[tid] = bv; ri[tid] = bi;
        __syncthreads();
        for (int s = 128; s > 0; s >>= 1) {
            if (tid < s) {
                const float ov = rv[tid + s]; const int oi = ri[tid + s];
                if (ov < rv[tid] || (ov == rv[tid] && oi < ri[tid])) { rv[tid] = ov; ri[tid] = oi; }
            }
            __syncthreads();
        }
        if (tid == 0) {
            const int j = ri[0];
            s_edges[edgeOffset + i * 16 + k] = j;
            d_edges[edgeOffset + i * 16 + k] = i;
            sd[j] = INF_F;
        }
        __syncthreads();
    }
}

// ---------------------------------------------------------------------------
// Feature kNN (k=127): distances via sq[i]+sq[j]-2*dot; 127 argmin passes.
// ---------------------------------------------------------------------------
__global__ void eb_knn_feat(const float* __restrict__ x1, const float* __restrict__ sq,
                            int M, int C, int* __restrict__ nn_feat)
{
    const int i = blockIdx.x;
    const int tid = threadIdx.x;
    __shared__ float sxi[128];
    __shared__ float sd[4096];
    __shared__ float rv[256];
    __shared__ int   ri[256];
    for (int c = tid; c < C; c += 256) sxi[c] = x1[(long)i * C + c];
    __syncthreads();
    const float sqi = sq[i];
    for (int j = tid; j < M; j += 256) {
        const float* __restrict__ xj = x1 + (long)j * C;
        __builtin_prefetch(xj + C, 0, 1);
        float dot = 0.f;
        for (int c = 0; c < C; ++c) dot += sxi[c] * xj[c];
        sd[j] = (j == i) ? INF_F : (sqi + sq[j] - 2.0f * dot);
    }
    __syncthreads();
    for (int k = 0; k < 127; ++k) {
        float bv = INF_F; int bi = M;
        for (int j = tid; j < M; j += 256)
            if (sd[j] < bv) { bv = sd[j]; bi = j; }
        rv[tid] = bv; ri[tid] = bi;
        __syncthreads();
        for (int s = 128; s > 0; s >>= 1) {
            if (tid < s) {
                const float ov = rv[tid + s]; const int oi = ri[tid + s];
                if (ov < rv[tid] || (ov == rv[tid] && oi < ri[tid])) { rv[tid] = ov; ri[tid] = oi; }
            }
            __syncthreads();
        }
        if (tid == 0) {
            nn_feat[i * 127 + k] = ri[0];
            sd[ri[0]] = INF_F;
        }
        __syncthreads();
    }
}

// ---------------------------------------------------------------------------
// Deterministic stateless RNG (stand-in for the JAX threefry stream).
// ---------------------------------------------------------------------------
__device__ __forceinline__ unsigned eb_hash(unsigned x)
{
    x ^= x >> 16; x *= 0x7feb352dU;
    x ^= x >> 15; x *= 0x846ca68bU;
    x ^= x >> 16; return x;
}
__device__ __forceinline__ float eb_u01(unsigned h) { return (h >> 8) * (1.0f / 16777216.0f); }

__global__ void eb_perturb(float* __restrict__ emb, int M, int ldE)
{
    const int total = M * 20;
    for (int t = blockIdx.x * blockDim.x + threadIdx.x; t < total; t += gridDim.x * blockDim.x) {
        const int i = t / 20, c = t - i * 20;
        emb[(long)i * ldE + c] += eb_u01(eb_hash((unsigned)t * 2654435761u ^ 0xA511E9B3u)) * 1e-4f;
    }
}

// ---------------------------------------------------------------------------
// Gumbel top-16 over the 127 feature-kNN candidates per row.
// ---------------------------------------------------------------------------
__global__ void eb_gumbel_topk(const float* __restrict__ emb, int ldE,
                               const int* __restrict__ nn_feat, int M,
                               int* __restrict__ s_edges, int* __restrict__ d_edges)
{
    const int i = blockIdx.x;
    const int tid = threadIdx.x; // 128 threads
    __shared__ float se[20];
    __shared__ float sc[128];
    __shared__ float rv[128];
    __shared__ int   ri[128];
    if (tid < 20) se[tid] = emb[(long)i * ldE + tid];
    __syncthreads();
    if (tid < 127) {
        const int src = nn_feat[i * 127 + tid];
        float d2 = 0.f;
        for (int c = 0; c < 20; ++c) {
            const float df = emb[(long)src * ldE + c] - se[c];
            d2 += df * df;
        }
        const float p = __expf(-d2);
        const float u = eb_u01(eb_hash((unsigned)(i * 127 + tid) * 0x9E3779B9u ^ 0xC0FFEE11u));
        const float g = -logf(-logf(u + 1e-20f) + 1e-20f);
        sc[tid] = logf(p + 1e-20f) + g;
    } else {
        sc[tid] = -INF_F;
    }
    __syncthreads();
    for (int k = 0; k < 16; ++k) {
        rv[tid] = sc[tid]; ri[tid] = tid;
        __syncthreads();
        for (int s = 64; s > 0; s >>= 1) {
            if (tid < s) {
                const float ov = rv[tid + s]; const int oi = ri[tid + s];
                if (ov > rv[tid] || (ov == rv[tid] && oi < ri[tid])) { rv[tid] = ov; ri[tid] = oi; }
            }
            __syncthreads();
        }
        if (tid == 0) {
            const int w = ri[0];
            s_edges[i * 16 + k] = nn_feat[i * 127 + w];
            d_edges[i * 16 + k] = i;
            sc[w] = -INF_F;
        }
        __syncthreads();
    }
}

// ---------------------------------------------------------------------------
// alpha0[e,c] = a_dst[d,c] - a_src[s,c] + (pos1[d]-pos1[s]) . Wp[:,c] + bp[c]
// ---------------------------------------------------------------------------
__global__ void eb_alpha0(const float* __restrict__ a_src, const float* __restrict__ a_dst,
                          const float* __restrict__ pos1,
                          const float* __restrict__ Wp, const float* __restrict__ bp,
                          const int* __restrict__ s_e, const int* __restrict__ d_e,
                          int E, int C, float* __restrict__ alpha0)
{
    const int tot = E * C;
    for (int t = blockIdx.x * blockDim.x + threadIdx.x; t < tot; t += gridDim.x * blockDim.x) {
        const int e = t / C, c = t - e * C;
        const int s = s_e[e], d = d_e[e];
        const float dx = pos1[d * 3 + 0] - pos1[s * 3 + 0];
        const float dy = pos1[d * 3 + 1] - pos1[s * 3 + 1];
        const float dz = pos1[d * 3 + 2] - pos1[s * 3 + 2];
        const float delta = dx * Wp[c] + dy * Wp[C + c] + dz * Wp[2 * C + c] + bp[c];
        alpha0[t] = a_dst[(long)d * C + c] - a_src[(long)s * C + c] + delta;
    }
}

// ---------------------------------------------------------------------------
// Per-target softmax + weighted aggregation: each target owns exactly 32
// edge slots in two contiguous runs of 16 -> no atomics, register-resident.
// ---------------------------------------------------------------------------
__global__ void eb_aggregate(const float* __restrict__ alpha2, const float* __restrict__ vfeat,
                             const float* __restrict__ pos1,
                             const float* __restrict__ Wp, const float* __restrict__ bp,
                             const int* __restrict__ s_e, int M, int C, float* __restrict__ out)
{
    const int i = blockIdx.x;
    const int c = threadIdx.x; // C == 128 == blockDim.x
    __shared__ float sdp[32][3];
    __shared__ int   ss[32];
    if (c < 32) {
        const int e = (c < 16) ? (i * 16 + c) : (M * 16 + i * 16 + (c - 16));
        const int s = s_e[e];
        ss[c] = s;
        sdp[c][0] = pos1[i * 3 + 0] - pos1[s * 3 + 0];
        sdp[c][1] = pos1[i * 3 + 1] - pos1[s * 3 + 1];
        sdp[c][2] = pos1[i * 3 + 2] - pos1[s * 3 + 2];
    }
    __syncthreads();
    const float w0 = Wp[c], w1 = Wp[C + c], w2 = Wp[2 * C + c], bpc = bp[c];

    float av[32];
    float amax = -INF_F;
    #pragma unroll
    for (int t = 0; t < 32; ++t) {
        const int e = (t < 16) ? (i * 16 + t) : (M * 16 + i * 16 + (t - 16));
        const float a = alpha2[(long)e * C + c];
        av[t] = a;
        amax = fmaxf(amax, a);
    }
    float ssum = 0.f;
    #pragma unroll
    for (int t = 0; t < 32; ++t) { av[t] = __expf(av[t] - amax); ssum += av[t]; }
    const float inv = 1.0f / (ssum + 1e-16f);
    float acc = 0.f;
    #pragma unroll
    for (int t = 0; t < 32; ++t) {
        const float delta = sdp[t][0] * w0 + sdp[t][1] * w1 + sdp[t][2] * w2 + bpc;
        acc += av[t] * inv * (vfeat[(long)ss[t] * C + c] + delta);
    }
    out[(long)i * C + c] = acc;
}

// ---------------------------------------------------------------------------
// Host orchestration
// ---------------------------------------------------------------------------
extern "C" void kernel_launch(void* const* d_in, const int* in_sizes, int n_in,
                              void* d_out, int out_size, void* d_ws, size_t ws_size,
                              hipStream_t stream)
{
    (void)in_sizes; (void)n_in; (void)out_size; (void)ws_size;
    const int N = 8192, CIN = 64, C = 128, M = 4096, NE = N * 16, E = M * 32;

    const float* x      = (const float*)d_in[0];
    const float* pos    = (const float*)d_in[1];
    const int*   ei     = (const int*)  d_in[2];
    const float* Wd     = (const float*)d_in[3];
    const float* bd     = (const float*)d_in[4];
    const float* gd     = (const float*)d_in[5];
    const float* betd   = (const float*)d_in[6];
    const float* Wg1    = (const float*)d_in[7];
    const float* bg1    = (const float*)d_in[8];
    const float* gg     = (const float*)d_in[9];
    const float* betg   = (const float*)d_in[10];
    const float* Wg2    = (const float*)d_in[11];
    const float* bg2    = (const float*)d_in[12];
    const float* Wlin   = (const float*)d_in[13];
    const float* Wsrc   = (const float*)d_in[14];
    const float* Wdst   = (const float*)d_in[15];
    const float* Wp     = (const float*)d_in[16];
    const float* bp     = (const float*)d_in[17];
    const float* Wa1    = (const float*)d_in[18];
    const float* ba1    = (const float*)d_in[19];
    const float* ga     = (const float*)d_in[20];
    const float* beta_a = (const float*)d_in[21];
    const float* Wa2    = (const float*)d_in[22];
    const float* ba2    = (const float*)d_in[23];
    const float* Wu     = (const float*)d_in[24];
    const float* bu     = (const float*)d_in[25];

    // ---- workspace carve-out (fits MI455X's 192MB global L2) ----
    size_t off = 0;
    auto alloc = [&](size_t bytes) -> void* {
        void* p = (char*)d_ws + off;
        off += (bytes + 255) & ~(size_t)255;
        return p;
    };
    float* h       = (float*)alloc((size_t)N * C * 4);
    float* pooled  = (float*)alloc((size_t)N * C * 4);
    float* meanb   = (float*)alloc(C * 4);
    float* istdb   = (float*)alloc(C * 4);
    int*   fpsidx  = (int*)  alloc((size_t)M * 4);
    float* x1      = (float*)alloc((size_t)M * C * 4);
    float* pos1    = (float*)alloc((size_t)M * 3 * 4);
    float* sq      = (float*)alloc((size_t)M * 4);
    int*   nn_feat = (int*)  alloc((size_t)M * 127 * 4);
    float* hg      = (float*)alloc((size_t)M * C * 4);
    float* emb     = (float*)alloc((size_t)M * 64 * 4);      // padded ld = 64
    int*   s_edges = (int*)  alloc((size_t)E * 4);
    int*   d_edges = (int*)  alloc((size_t)E * 4);
    float* a_src   = (float*)alloc((size_t)M * C * 4);
    float* a_dst   = (float*)alloc((size_t)M * C * 4);
    float* vfeat   = (float*)alloc((size_t)M * C * 4);
    f16*   Bsw     = (f16*)  alloc((size_t)128 * 128 * 2);   // swizzled weights (reused)
    float* biasPad = (float*)alloc(128 * 4);                 // padded bias (reused)
    float* alpha0  = (float*)alloc((size_t)E * C * 4);       // 64 MB
    float* alpha1  = (float*)alloc((size_t)E * C * 4);       // 64 MB
    float* aggout  = (float*)alloc((size_t)M * C * 4);

    auto launch_gemm = [&](const float* A, const float* B, const float* bias,
                           const float* resid, float* D, int Mm, int Npad, int Nact, int K) {
        const int swTotal = (Npad / 16) * (K / 32) * 32 * 16;
        eb_swizzleB<<<std::min((swTotal + 255) / 256, 256), 256, 0, stream>>>(
            B, bias, K, Nact, Npad, Nact, Bsw, biasPad);
        const int tasks = (Mm / 16) * (Npad / 64);
        const int blocks = std::max(1, std::min((tasks + 7) / 8, 4096));
        eb_gemm_wmma<<<blocks, 256, 0, stream>>>(A, Bsw, biasPad, resid, D,
                                                 Mm, Npad, K, K, Npad);
    };
    auto ew_blocks = [&](int total) { return std::min((total + 255) / 256, 4096); };

    // 1) h = relu(bn(x @ Wd + bd))
    launch_gemm(x, Wd, bd, nullptr, h, N, C, C, CIN);
    eb_bn_stats<<<C, 256, 0, stream>>>(h, N, C, meanb, istdb);
    eb_bn_apply_relu<<<ew_blocks(N * C), 256, 0, stream>>>(h, N * C, C, meanb, istdb, gd, betd);

    // 2) neighbor max-pool with self loop
    hipMemcpyAsync(pooled, h, (size_t)N * C * 4, hipMemcpyDeviceToDevice, stream);
    eb_edge_max<<<ew_blocks(NE * C), 256, 0, stream>>>(h, ei, NE, C, pooled);

    // 3) FPS + gather
    eb_fps<<<1, 1024, 0, stream>>>(pos, N, M, fpsidx);
    eb_gather<<<ew_blocks(M * C), 256, 0, stream>>>(pooled, pos, fpsidx, M, C, x1, pos1);

    // 4) graph generation
    eb_rowsq<<<ew_blocks(M), 256, 0, stream>>>(x1, M, C, sq);
    eb_knn_feat<<<M, 256, 0, stream>>>(x1, sq, M, C, nn_feat);
    eb_knn_pos<<<M, 256, 0, stream>>>(pos1, M, s_edges, d_edges, M * 16);

    launch_gemm(x1, Wg1, bg1, nullptr, hg, M, C, C, C);
    eb_bn_stats<<<C, 256, 0, stream>>>(hg, M, C, meanb, istdb);
    eb_bn_apply_relu<<<ew_blocks(M * C), 256, 0, stream>>>(hg, M * C, C, meanb, istdb, gg, betg);
    launch_gemm(hg, Wg2, bg2, nullptr, emb, M, 64, 20, C);   // emb padded to ld 64
    eb_perturb<<<ew_blocks(M * 20), 256, 0, stream>>>(emb, M, 64);
    eb_gumbel_topk<<<M, 128, 0, stream>>>(emb, 64, nn_feat, M, s_edges, d_edges);

    // 5) PointTransformerConv
    launch_gemm(x1, Wsrc, nullptr, nullptr, a_src, M, C, C, C);
    launch_gemm(x1, Wdst, nullptr, nullptr, a_dst, M, C, C, C);
    launch_gemm(x1, Wlin, nullptr, nullptr, vfeat, M, C, C, C);
    eb_alpha0<<<ew_blocks(E * C), 256, 0, stream>>>(a_src, a_dst, pos1, Wp, bp,
                                                    s_edges, d_edges, E, C, alpha0);
    launch_gemm(alpha0, Wa1, ba1, nullptr, alpha1, E, C, C, C);
    eb_bn_stats<<<C, 256, 0, stream>>>(alpha1, E, C, meanb, istdb);
    eb_bn_apply_relu<<<ew_blocks(E * C), 256, 0, stream>>>(alpha1, E * C, C, meanb, istdb, ga, beta_a);
    launch_gemm(alpha1, Wa2, ba2, nullptr, alpha0, E, C, C, C);   // alpha2 -> alpha0 buffer

    eb_aggregate<<<M, 128, 0, stream>>>(alpha0, vfeat, pos1, Wp, bp, s_edges, M, C, aggout);

    // 6) final: out @ Wu + bu + x1
    launch_gemm(aggout, Wu, bu, x1, (float*)d_out, M, C, C, C);
}
